// LossFunction_40346922778859
// MI455X (gfx1250) — matrix operand order repaired and checked by reading, not compiled, not run on hardware
//
#include <hip/hip_runtime.h>
#include <hip/hip_fp16.h>
#include <math.h>

typedef __attribute__((ext_vector_type(16))) _Float16 v16h;
typedef __attribute__((ext_vector_type(8)))  _Float16 v8h;
typedef __attribute__((ext_vector_type(8)))  float    v8f;

#define EMBED      192
#define NROWS      8192
#define HALF_N     4096
#define LDS_STRIDE 200   // halves per LDS row (192 + 8 pad to break bank conflicts)
#define NTILE      (NROWS / 64)              // 128 tiles per dim
#define NTRI       (NTILE * (NTILE + 1) / 2) // 8256 lower-triangle blocks

// workspace layout (bytes)
#define OFF_TOTALH 0
#define OFF_SQ     (NROWS * EMBED * 2)          // f16 staged "total" matrix, 3 MB
#define OFF_COLSUM (OFF_SQ + NROWS * 4)
#define OFF_SUMSQ  (OFF_COLSUM + EMBED * 4)
#define OFF_C0     (OFF_SUMSQ + 4)
#define OFF_LOSS   (OFF_C0 + 4)

// Pass 1: per-row squared norms, column sums, f16 conversion into "total" order.
// total[i] = x[2i] for i<4096 ; total[4096+i] = x[2i+1]
__global__ __launch_bounds__(256) void prep_kernel(
    const float* __restrict__ x, _Float16* __restrict__ totalH,
    float* __restrict__ sq, float* __restrict__ colsum, float* __restrict__ sumSq) {
  __shared__ float colsumLoc[EMBED];
  __shared__ float partials[256];
  int tid = threadIdx.x;
  if (tid < EMBED) colsumLoc[tid] = 0.f;
  __syncthreads();

  int rowInBlk = tid >> 1;
  int half = tid & 1;
  int row = blockIdx.x * 128 + rowInBlk;
  int trow = (row & 1) ? (HALF_N + (row >> 1)) : (row >> 1);
  const float* xr = x + (long)row * EMBED;
  _Float16* tr = totalH + (long)trow * EMBED;

  float p = 0.f;
  int k0 = half * 96;
  for (int k = k0; k < k0 + 96; ++k) {
    float v = xr[k];
    p += v * v;
    tr[k] = (_Float16)v;
    atomicAdd(&colsumLoc[k], v);
  }
  partials[tid] = p;
  __syncthreads();
  if (half == 0) {
    float s = partials[tid] + partials[tid + 1];
    sq[row] = s;
    atomicAdd(sumSq, s);
  }
  __syncthreads();
  if (tid < EMBED) atomicAdd(&colsum[tid], colsumLoc[tid]);
}

// Pass 2: bandwidth from the identity sum(L2) = 2n*sum||x||^2 - 2*||colsum||^2
__global__ void bw_kernel(const float* __restrict__ colsum,
                          const float* __restrict__ sumSq, float* __restrict__ c0out) {
  __shared__ float red[256];
  int tid = threadIdx.x;
  float v = (tid < EMBED) ? colsum[tid] : 0.f;
  red[tid] = v * v;
  __syncthreads();
  for (int s = 128; s > 0; s >>= 1) {
    if (tid < s) red[tid] += red[tid + s];
    __syncthreads();
  }
  if (tid == 0) {
    double n = (double)NROWS;
    double sumL2 = 2.0 * n * (double)(*sumSq) - 2.0 * (double)red[0];
    double bw = sumL2 / (n * n - n) / 4.0;  // / KERNEL_MUL^(KERNEL_NUM//2) = 2^2
    *c0out = (float)(-1.0 / bw);
  }
}

// Pass 3: fused Gram-tile (WMMA f16->f32) + multi-kernel RBF + signed reduction.
// Lower-triangle 64x64 tiles; block = 128 threads (4 wave32), wave = 32x32 region.
__global__ __launch_bounds__(128) void mmd_kernel(
    const _Float16* __restrict__ totalH, const float* __restrict__ sq,
    const float* __restrict__ c0ptr, float* __restrict__ lossAcc) {
  __shared__ __attribute__((aligned(32))) _Float16 lds[128 * LDS_STRIDE];
  __shared__ float sqA[64], sqB[64];
  __shared__ float red[128];

  // Map linear block id -> lower-triangle tile (by >= bx).
  int b = blockIdx.x;
  int by = (int)(sqrtf(2.0f * (float)b + 0.25f) - 0.5f);
  while ((by + 1) * (by + 2) / 2 <= b) ++by;
  while (by * (by + 1) / 2 > b) --by;
  int bx = b - by * (by + 1) / 2;
  int rowBase = by * 64;   // i-tile (rows of A)
  int colBase = bx * 64;   // j-tile (rows used as B columns)

  int tid = threadIdx.x;

  // Stage 64 A-rows (lds rows 0..63) + 64 B-rows (64..127) as f16, padded stride.
  unsigned int* ldsU = (unsigned int*)lds;
  const unsigned int* gU = (const unsigned int*)totalH;
  for (int idx = tid; idx < 128 * 96; idx += 128) {
    int r = idx / 96, c = idx - r * 96;
    int grow = (r < 64) ? (rowBase + r) : (colBase + (r - 64));
    ldsU[r * (LDS_STRIDE / 2) + c] = gU[(long)grow * 96 + c];
  }
  if (tid < 64) sqA[tid] = sq[rowBase + tid];
  else          sqB[tid - 64] = sq[colBase + (tid - 64)];
  __syncthreads();

  float cu = (*c0ptr) * 0.0625f;     // -1/(16*bw): u = exp(L2*cu); terms u,u2,u4,u8,u16
  int lane = tid & 31;
  int wave = tid >> 5;
  int m0 = (wave >> 1) * 32;         // wave rows: 0 or 32
  int n0 = (wave & 1) * 32;          // wave cols: 0 or 32
  int m  = lane & 15;
  int hi = lane >> 4;

  v8f acc00 = {}, acc01 = {}, acc10 = {}, acc11 = {};
  #pragma unroll
  for (int kb = 0; kb < EMBED; kb += 32) {
    // A fragment (16-bit 16x32): lo lanes K=[kb..kb+7],[kb+16..kb+23]; hi lanes +8.
    const v8h* a0lo = (const v8h*)&lds[(m0 + m)      * LDS_STRIDE + kb + hi * 8];
    const v8h* a0hi = (const v8h*)&lds[(m0 + m)      * LDS_STRIDE + kb + 16 + hi * 8];
    const v8h* a1lo = (const v8h*)&lds[(m0 + 16 + m) * LDS_STRIDE + kb + hi * 8];
    const v8h* a1hi = (const v8h*)&lds[(m0 + 16 + m) * LDS_STRIDE + kb + 16 + hi * 8];
    // B fragment (16-bit 32x16): lo lanes K=[kb..kb+7],[kb+16..kb+23]; hi lanes +8.
    const v8h* b0lo = (const v8h*)&lds[(64 + n0 + m)      * LDS_STRIDE + kb + hi * 8];
    const v8h* b0hi = (const v8h*)&lds[(64 + n0 + m)      * LDS_STRIDE + kb + 16 + hi * 8];
    const v8h* b1lo = (const v8h*)&lds[(64 + n0 + 16 + m) * LDS_STRIDE + kb + hi * 8];
    const v8h* b1hi = (const v8h*)&lds[(64 + n0 + 16 + m) * LDS_STRIDE + kb + 16 + hi * 8];

    v16h a0 = __builtin_shufflevector(*a0lo, *a0hi, 0,1,2,3,4,5,6,7,8,9,10,11,12,13,14,15);
    v16h a1 = __builtin_shufflevector(*a1lo, *a1hi, 0,1,2,3,4,5,6,7,8,9,10,11,12,13,14,15);
    v16h b0 = __builtin_shufflevector(*b0lo, *b0hi, 0,1,2,3,4,5,6,7,8,9,10,11,12,13,14,15);
    v16h b1 = __builtin_shufflevector(*b1lo, *b1hi, 0,1,2,3,4,5,6,7,8,9,10,11,12,13,14,15);

    acc00 = __builtin_amdgcn_wmma_f32_16x16x32_f16(false, a0, false, b0, (short)0, acc00, false, false);
    acc01 = __builtin_amdgcn_wmma_f32_16x16x32_f16(false, a0, false, b1, (short)0, acc01, false, false);
    acc10 = __builtin_amdgcn_wmma_f32_16x16x32_f16(false, a1, false, b0, (short)0, acc10, false, false);
    acc11 = __builtin_amdgcn_wmma_f32_16x16x32_f16(false, a1, false, b1, (short)0, acc11, false, false);
  }

  // Epilogue: L2 -> u=exp(L2*cu) -> u+u^2+u^4+u^8+u^16 (1 exp + 4 squarings).
  float local = 0.f;
  int nlane = m;  // N = lane%16 in C/D layout
  #pragma unroll
  for (int t = 0; t < 4; ++t) {
    v8f acc = (t == 0) ? acc00 : (t == 1) ? acc01 : (t == 2) ? acc10 : acc11;
    int jl = n0 + ((t & 1) ? 16 : 0) + nlane;
    float sqj = sqB[jl];
    #pragma unroll
    for (int e = 0; e < 8; ++e) {
      int il = m0 + ((t >> 1) ? 16 : 0) + e + 8 * hi;   // M = e + 8*hi
      float l2 = fmaxf(sqA[il] + sqj - 2.f * acc[e], 0.f);
      float u  = __expf(l2 * cu);
      float u2 = u * u, u4 = u2 * u2, u8 = u4 * u4, u16 = u8 * u8;
      local += ((u + u2) + (u4 + u8)) + u16;
    }
  }

  red[tid] = local;
  __syncthreads();
  for (int s = 64; s > 0; s >>= 1) {
    if (tid < s) red[tid] += red[tid + s];
    __syncthreads();
  }
  if (tid == 0) {
    // Block-uniform quadrant sign (tiles never straddle the 4096 boundary),
    // off-diagonal tiles counted twice via symmetry.
    float sgn = ((rowBase < HALF_N) == (colBase < HALF_N)) ? 1.f : -1.f;
    float wt  = (bx == by) ? 1.f : 2.f;
    atomicAdd(lossAcc, red[0] * sgn * wt);
  }
}

__global__ void finalize_kernel(const float* __restrict__ lossAcc, float* __restrict__ out) {
  if (threadIdx.x == 0)
    out[0] = *lossAcc * (1.0f / ((float)HALF_N * (float)HALF_N));
}

extern "C" void kernel_launch(void* const* d_in, const int* in_sizes, int n_in,
                              void* d_out, int out_size, void* d_ws, size_t ws_size,
                              hipStream_t stream) {
  const float* x = (const float*)d_in[0];   // (8192, 192) f32
  char* ws = (char*)d_ws;
  _Float16* totalH = (_Float16*)(ws + OFF_TOTALH);
  float* sq      = (float*)(ws + OFF_SQ);
  float* colsum  = (float*)(ws + OFF_COLSUM);
  float* sumSq   = (float*)(ws + OFF_SUMSQ);
  float* c0      = (float*)(ws + OFF_C0);
  float* lossAcc = (float*)(ws + OFF_LOSS);

  // zero colsum + sumSq + c0 + lossAcc (contiguous)
  hipMemsetAsync(ws + OFF_COLSUM, 0, (EMBED + 3) * sizeof(float), stream);

  prep_kernel<<<64, 256, 0, stream>>>(x, totalH, sq, colsum, sumSq);
  bw_kernel<<<1, 256, 0, stream>>>(colsum, sumSq, c0);
  mmd_kernel<<<NTRI, 128, 0, stream>>>(totalH, sq, c0, lossAcc);
  finalize_kernel<<<1, 64, 0, stream>>>(lossAcc, (float*)d_out);
}
